// MaskedConvFlow_23184233464384
// MI455X (gfx1250) — compile-verified, matrix operand order checked by reading, not run on hardware
//
#include <hip/hip_runtime.h>

// ---------------- problem constants (from reference) ----------------
#define NB   32      // batch
#define CIN  64
#define HH   64
#define WW   64
#define CHID 256
#define COUT 128
#define K1   384     // CIN*KH*KW = 64*2*3 (im2col K for conv1)
#define K2   256     // CHID (K for 1x1 conv)

typedef __attribute__((ext_vector_type(16))) __bf16          v16bf;
typedef __attribute__((ext_vector_type(8)))  float           v8f;
typedef __attribute__((ext_vector_type(4)))  float           f4;
typedef __attribute__((ext_vector_type(4)))  int             v4i;
typedef __attribute__((ext_vector_type(8)))  unsigned short  su8;

#if defined(__has_builtin)
#  if __has_builtin(__builtin_amdgcn_global_load_async_to_lds_b128)
#    define HAVE_ASYNC_LDS 1
#  endif
#endif
#ifndef HAVE_ASYNC_LDS
#  define HAVE_ASYNC_LDS 0
#endif

union BFrag { su8 h[2]; v16bf v; };

static __device__ __forceinline__ unsigned short f2bf(float f) {
  unsigned int u = __float_as_uint(f);
  u += 0x7FFFu + ((u >> 16) & 1u);          // round-to-nearest-even
  return (unsigned short)(u >> 16);
}

// Load a 16x32 bf16 A/B fragment slice for this lane.
// p must point at (row * stride + k0 + (lane>>4)*8) in a bf16 [row][K] array.
// Layout per ISA 7.12.2: lanes<16 hold K {0..7,16..23}, lanes>=16 hold {8..15,24..31}.
static __device__ __forceinline__ v16bf frag_from(const unsigned short* p) {
  BFrag f;
  f.h[0] = *(const su8*)(p);        // K group 0 (8 bf16, 16B)
  f.h[1] = *(const su8*)(p + 16);   // K group 1 (8 bf16, 16B)
  return f.v;
}

union SMem {
  unsigned short B[64 * K1];   // im2col panel  [n][k]  bf16  (48 KB)
  unsigned short C[64 * K2];   // relu(hidden)  [n][k]  bf16  (32 KB)
  float          P[COUT * WW]; // conv2 output  [m][n]  f32   (32 KB)
  float          red[256];     // logdet reduction scratch
};

// ---------------- prep: f32 weights -> bf16 in workspace, zero logdet ----------------
__global__ void mcf_prep(const float* __restrict__ Wm, const float* __restrict__ W1,
                         unsigned short* __restrict__ WmB, unsigned short* __restrict__ W1B,
                         float* __restrict__ logdet)
{
  int idx    = blockIdx.x * blockDim.x + threadIdx.x;
  int stride = gridDim.x * blockDim.x;
  for (int i = idx; i < CHID * K1; i += stride) WmB[i] = f2bf(Wm[i]);
  for (int i = idx; i < COUT * K2; i += stride) W1B[i] = f2bf(W1[i]);
  if (idx < NB) logdet[idx] = 0.f;
}

// ---------------- fused masked-conv -> relu -> 1x1 -> affine flow ----------------
__global__ __launch_bounds__(256)
void mcf_main(const float* __restrict__ x,
              const float* __restrict__ bm,
              const float* __restrict__ b1,
              const unsigned short* __restrict__ WmB,   // [256][384] bf16
              const unsigned short* __restrict__ W1B,   // [128][256] bf16
              float* __restrict__ out,
              float* __restrict__ logdet)
{
  __shared__ SMem  sm;
  __shared__ float stage[CIN * 2 * WW];   // x rows (h-2, h-1), f32 (32 KB)

  const int tid  = threadIdx.x;       // 0..255, 8 wave32s
  const int wave = tid >> 5;
  const int lane = tid & 31;
  const int ln16 = lane & 15;
  const int half = lane >> 4;

  const int bh = blockIdx.x;
  const int b  = bh >> 6;             // / HH
  const int h  = bh & (HH - 1);

  // Prefetch the x row this block re-reads in the epilogue (16 KB -> global_prefetch_b8)
  {
    const float* px = x + (((size_t)b * CIN + (tid >> 2)) * HH + h) * WW + (tid & 3) * 16;
    __builtin_prefetch(px, 0, 1);
  }

  // ---- Phase 0: async-stage x[b, :, h-2:h, :] into LDS (global -> LDS, no VGPR hop).
  // 2048 chunks of 16 B; rows above the image are zero-filled with DS stores.
  #pragma unroll
  for (int i = 0; i < 8; ++i) {
    int q   = tid + i * 256;
    int ci  = q >> 5;                 // 32 chunks per channel (2 rows x 16)
    int rem = q & 31;
    int r   = rem >> 4;               // causal row select: h-2+r
    int cw  = rem & 15;               // 16-B chunk within the 64-float row
    int row = h - 2 + r;
    float* dst = stage + (ci * 2 + r) * WW + cw * 4;
    if (row >= 0) {
      const float* src = x + (((size_t)b * CIN + ci) * HH + row) * WW + cw * 4;
#if HAVE_ASYNC_LDS
      __builtin_amdgcn_global_load_async_to_lds_b128((v4i*)src, (v4i*)dst, 0, 0);
#else
      *(f4*)dst = *(const f4*)src;
#endif
    } else {
      *(f4*)dst = (f4){0.f, 0.f, 0.f, 0.f};
    }
  }
#if HAVE_ASYNC_LDS
#  if __has_builtin(__builtin_amdgcn_s_wait_asynccnt)
  __builtin_amdgcn_s_wait_asynccnt(0);
#  else
  asm volatile("s_wait_asynccnt 0x0" ::: "memory");
#  endif
#endif
  __syncthreads();

  // ---- Phase 1: im2col -> sm.B[n][k] bf16, k = ci*6 + kh*3 + kw (matches Wm flat order)
  for (int e = tid; e < 64 * K1; e += 256) {
    int n  = e / K1, k = e - n * K1;
    int ci = k / 6,  rem = k - ci * 6;
    int kh = rem / 3, kw = rem - kh * 3;
    int col = n + kw - 1;              // width halo; rows already staged/zeroed
    float v = 0.f;
    if ((unsigned)col < WW)
      v = stage[(ci * 2 + kh) * WW + col];
    sm.B[e] = f2bf(v);
  }
  __syncthreads();

  // ---- Phase 2: GEMM1  c[256x64] = Wm[256x384] @ B[384x64], f32 accumulate
  v8f acc1[2][4];
  #pragma unroll
  for (int t = 0; t < 2; ++t)
    #pragma unroll
    for (int nt = 0; nt < 4; ++nt)
      acc1[t][nt] = (v8f){0.f,0.f,0.f,0.f,0.f,0.f,0.f,0.f};

  const int m0 = wave * 32;            // this wave's 32 hidden channels
  for (int kc = 0; kc < K1 / 32; ++kc) {
    const int k0 = kc * 32;
    v16bf a0 = frag_from(WmB + (size_t)(m0      + ln16) * K1 + k0 + half * 8);
    v16bf a1 = frag_from(WmB + (size_t)(m0 + 16 + ln16) * K1 + k0 + half * 8);
    #pragma unroll
    for (int nt = 0; nt < 4; ++nt) {
      v16bf bf = frag_from(sm.B + (nt * 16 + ln16) * K1 + k0 + half * 8);
      acc1[0][nt] = __builtin_amdgcn_wmma_f32_16x16x32_bf16(false, a0, false, bf,
                      (short)0, acc1[0][nt], false, false);
      acc1[1][nt] = __builtin_amdgcn_wmma_f32_16x16x32_bf16(false, a1, false, bf,
                      (short)0, acc1[1][nt], false, false);
    }
  }
  __syncthreads();   // sm.B reads complete before sm.C (aliased) is written

  // ---- Phase 3: bias + ReLU, repack accumulators to sm.C[n][k] bf16.
  // C/D layout: VGPR r holds M = tile + half*8 + r at N = ln16 -> 8 contiguous K values.
  #pragma unroll
  for (int t = 0; t < 2; ++t) {
    const int mb = m0 + t * 16 + half * 8;
    #pragma unroll
    for (int nt = 0; nt < 4; ++nt) {
      unsigned short pk[8];
      #pragma unroll
      for (int r = 0; r < 8; ++r) {
        float v = acc1[t][nt][r] + bm[mb + r];
        pk[r] = f2bf(v > 0.f ? v : 0.f);
      }
      *(su8*)(sm.C + (nt * 16 + ln16) * K2 + mb) = *(const su8*)pk;   // ds_store_b128
    }
  }
  __syncthreads();

  // ---- Phase 4: GEMM2  p[128x64] = W1[128x256] @ relu(c)[256x64]
  v8f acc2[4];
  #pragma unroll
  for (int nt = 0; nt < 4; ++nt)
    acc2[nt] = (v8f){0.f,0.f,0.f,0.f,0.f,0.f,0.f,0.f};

  const int m2 = wave * 16;            // this wave's 16 output rows of p
  for (int kc = 0; kc < K2 / 32; ++kc) {
    const int k0 = kc * 32;
    v16bf a = frag_from(W1B + (size_t)(m2 + ln16) * K2 + k0 + half * 8);
    #pragma unroll
    for (int nt = 0; nt < 4; ++nt) {
      v16bf bf = frag_from(sm.C + (nt * 16 + ln16) * K2 + k0 + half * 8);
      acc2[nt] = __builtin_amdgcn_wmma_f32_16x16x32_bf16(false, a, false, bf,
                   (short)0, acc2[nt], false, false);
    }
  }
  __syncthreads();   // sm.C reads complete before sm.P (aliased) is written

  #pragma unroll
  for (int nt = 0; nt < 4; ++nt)
    #pragma unroll
    for (int r = 0; r < 8; ++r)
      sm.P[(m2 + half * 8 + r) * WW + nt * 16 + ln16] = acc2[nt][r];
  __syncthreads();

  // ---- Phase 5: affine epilogue + logdet reduction
  float lsum = 0.f;
  for (int e = tid; e < CIN * WW; e += 256) {
    int c = e >> 6, n = e & (WW - 1);
    float mu = sm.P[c * WW + n] + b1[c];
    float ls = sm.P[(c + CIN) * WW + n] + b1[c + CIN] + 2.0f;
    float sc = 1.f / (1.f + __expf(-ls));          // sigmoid(log_scale + 2)
    size_t xi = (((size_t)b * CIN + c) * HH + h) * WW + n;
    out[xi] = sc * x[xi] + mu;
    lsum += __logf(sc);
  }
  __syncthreads();   // sm.P reads done before reduction scratch (aliased) is written
  sm.red[tid] = lsum;
  __syncthreads();
  #pragma unroll
  for (int s = 128; s > 0; s >>= 1) {
    if (tid < s) sm.red[tid] += sm.red[tid + s];
    __syncthreads();
  }
  if (tid == 0) atomicAdd(logdet + b, sm.red[0]);
}

// ---------------- launch ----------------
extern "C" void kernel_launch(void* const* d_in, const int* in_sizes, int n_in,
                              void* d_out, int out_size, void* d_ws, size_t ws_size,
                              hipStream_t stream) {
  const float* x  = (const float*)d_in[0];
  const float* Wm = (const float*)d_in[1];
  const float* bm = (const float*)d_in[2];
  const float* W1 = (const float*)d_in[3];
  const float* b1 = (const float*)d_in[4];

  float* out    = (float*)d_out;
  float* logdet = out + (size_t)NB * CIN * HH * WW;   // outputs concatenated flat

  unsigned short* WmB = (unsigned short*)d_ws;        // 192 KB bf16
  unsigned short* W1B = WmB + CHID * K1;              // +64 KB bf16

  mcf_prep<<<256, 256, 0, stream>>>(Wm, W1, WmB, W1B, logdet);
  mcf_main<<<NB * HH, 256, 0, stream>>>(x, bm, b1, WmB, W1B, out, logdet);
}
